// Attention_Layer_12395275616751
// MI455X (gfx1250) — compile-verified, matrix operand order checked by reading, not run on hardware
//
#include <hip/hip_runtime.h>
#include <hip/hip_bf16.h>
#include <math.h>

// ---------- types ----------
typedef __attribute__((ext_vector_type(16))) __bf16     v16bf;
typedef __attribute__((ext_vector_type(8)))  float      v8f;
typedef __attribute__((ext_vector_type(4)))  unsigned   u32x4;

struct BF16Pair { u32x4 lo, hi; };   // 32 bytes == v16bf

static __device__ __forceinline__ unsigned short f2bf(float f) {
    union { float f; unsigned u; } v; v.f = f;
    unsigned u = v.u;
    unsigned r = u + 0x7FFFu + ((u >> 16) & 1u);   // round-to-nearest-even
    return (unsigned short)(r >> 16);
}

// Native CDNA5 transcendental tanh (V_TANH_F32). One TRANS op instead of the
// ~25-VALU branchy libm expansion. v_nop covers the TRANS co-execution hazard
// (1 independent op required before the result is consumed).
static __device__ __forceinline__ float fast_tanh(float x) {
    float r;
    asm("v_tanh_f32 %0, %1\n\tv_nop" : "=v"(r) : "v"(x));
    return r;
}

#define B_  32
#define S_  2048
#define H_  512
#define SROW 520   // LDS row stride in bf16 elements (512 + 8 pad -> 4-bank rotate/row)

// ---------- kernel 0: W[k][n] -> Wt[n][k] bf16 ----------
__global__ __launch_bounds__(256) void convW(const float* __restrict__ W,
                                             unsigned short* __restrict__ Wt) {
    int idx = blockIdx.x * 256 + threadIdx.x;      // over H*H = 262144
    int n = idx >> 9, k = idx & 511;
    Wt[idx] = f2bf(W[k * H_ + n]);
}

// ---------- kernel 1: scores[b,s] = tanh(Y@W + b) . w  - 1000*(1-mask) ----------
__global__ __launch_bounds__(64) void scores_kernel(
    const float* __restrict__ Y, const float* __restrict__ maskY,
    const unsigned short* __restrict__ Wt,
    const float* __restrict__ bias, const float* __restrict__ wvec,
    float* __restrict__ scores) {

    __shared__ unsigned short ldsY[2][16 * SROW];   // 33,280 B: one 16x512 bf16 tile per wave

    const int wave = threadIdx.x >> 5;
    const int lane = threadIdx.x & 31;
    const int st   = blockIdx.x * 2 + wave;   // s-tile index, 16 rows each
    const int row0 = st * 16;                 // global row in flattened [B*S]
    unsigned short* my = ldsY[wave];

    // ---- stage 16 rows x 512 cols of Y (f32) into LDS as bf16, once ----
    const float4* Ysrc = (const float4*)(Y + (size_t)row0 * H_);
    #pragma unroll 4
    for (int it = 0; it < 64; ++it) {
        int e4 = it * 32 + lane;              // 0..2047 float4s
        int r  = e4 >> 7, c4 = e4 & 127;
        float4 v = Ysrc[r * 128 + c4];
        unsigned* dst = (unsigned*)&my[r * SROW + c4 * 4];
        dst[0] = (unsigned)f2bf(v.x) | ((unsigned)f2bf(v.y) << 16);
        dst[1] = (unsigned)f2bf(v.z) | ((unsigned)f2bf(v.w) << 16);
    }
    // same-wave DS store->load is kept in order by hardware (DScnt in-order), no barrier needed

    const int rA   = lane & 15;               // A row / B column within tile
    const int koff = (lane >> 4) << 3;        // lane half: K chunks {0..7,16..23} vs {8..15,24..31}

    float racc[8];
    #pragma unroll
    for (int v = 0; v < 8; ++v) racc[v] = 0.f;

    for (int nt = 0; nt < 32; ++nt) {         // 32 column tiles of 16 h
        const int h = nt * 16 + rA;
        v8f acc = {};
        const u32x4* bbase = (const u32x4*)&Wt[(size_t)h * H_];
        #pragma unroll 4
        for (int kt = 0; kt < 16; ++kt) {     // K = 512 in steps of 32
            const int kb = kt * 32 + koff;
            BF16Pair ta, tb;
            const u32x4* ap = (const u32x4*)&my[rA * SROW + kb];
            ta.lo = ap[0];                    // K run 0..7
            ta.hi = ap[2];                    // K run +16 (32 bytes further)
            const u32x4* bp = (const u32x4*)((const unsigned short*)bbase + kb);
            tb.lo = bp[0];
            tb.hi = bp[2];
            v16bf a  = __builtin_bit_cast(v16bf, ta);
            v16bf bm = __builtin_bit_cast(v16bf, tb);
            acc = __builtin_amdgcn_wmma_f32_16x16x32_bf16(
                false, a, false, bm, (short)0, acc, false, false);
        }
        const float bb = bias[h];
        const float ww = wvec[h];
        #pragma unroll
        for (int v = 0; v < 8; ++v)
            racc[v] += ww * fast_tanh(acc[v] + bb);   // column-partial of score per row
    }

    // reduce across the 16 lanes of each half (D layout: lane = N, VGPR v = row M / M+8)
    #pragma unroll
    for (int v = 0; v < 8; ++v) {
        float x = racc[v];
        x += __shfl_xor(x, 1);
        x += __shfl_xor(x, 2);
        x += __shfl_xor(x, 4);
        x += __shfl_xor(x, 8);
        racc[v] = x;
    }
    if ((lane & 15) == 0) {
        const int half = lane >> 4;           // lane 0 -> rows 0..7, lane 16 -> rows 8..15
        #pragma unroll
        for (int v = 0; v < 8; ++v) {
            int row = row0 + half * 8 + v;
            float m = maskY[row];
            scores[row] = racc[v] - 1000.f * (1.f - m);
        }
    }
}

// ---------- kernel 2: softmax over S per batch (in place) ----------
__global__ __launch_bounds__(256) void softmax_kernel(float* __restrict__ scores) {
    __shared__ float red[256];
    const int b = blockIdx.x, t = threadIdx.x;
    float* s = scores + (size_t)b * S_;

    float m = -1e30f;
    for (int i = t; i < S_; i += 256) m = fmaxf(m, s[i]);
    red[t] = m; __syncthreads();
    for (int o = 128; o > 0; o >>= 1) { if (t < o) red[t] = fmaxf(red[t], red[t + o]); __syncthreads(); }
    m = red[0]; __syncthreads();

    float sum = 0.f;
    for (int i = t; i < S_; i += 256) sum += expf(s[i] - m);
    red[t] = sum; __syncthreads();
    for (int o = 128; o > 0; o >>= 1) { if (t < o) red[t] += red[t + o]; __syncthreads(); }
    const float inv = 1.0f / red[0];

    for (int i = t; i < S_; i += 256) s[i] = expf(s[i] - m) * inv;
}

// ---------- kernel 3: partial weighted sums over S chunks ----------
__global__ __launch_bounds__(256) void wsum_partial(
    const float* __restrict__ Y, const float* __restrict__ alpha,
    float* __restrict__ part) {
    const int blk = blockIdx.x;               // B * 16 chunks
    const int b = blk >> 4, ch = blk & 15;
    const int t = threadIdx.x;
    const int h0 = t, h1 = t + 256;
    const float* Yb = Y + (size_t)b * S_ * H_;
    const float* al = alpha + (size_t)b * S_;
    float a0 = 0.f, a1 = 0.f;
    const int sbase = ch * 128;
    for (int i = 0; i < 128; ++i) {
        const int s = sbase + i;
        const float a = al[s];
        const float* row = Yb + (size_t)s * H_;
        a0 += a * row[h0];
        a1 += a * row[h1];
    }
    float* p = part + ((size_t)ch * B_ + b) * H_;
    p[h0] = a0;
    p[h1] = a1;
}

// ---------- kernel 4: fixed-order reduction of partials -> out[b,h] ----------
__global__ __launch_bounds__(256) void wsum_reduce(
    const float* __restrict__ part, float* __restrict__ out) {
    const int idx = blockIdx.x * 256 + threadIdx.x;   // over B*H = 16384
    float s = 0.f;
    #pragma unroll
    for (int c = 0; c < 16; ++c)
        s += part[(size_t)c * B_ * H_ + idx];
    out[idx] = s;
}

// ---------- launch ----------
extern "C" void kernel_launch(void* const* d_in, const int* in_sizes, int n_in,
                              void* d_out, int out_size, void* d_ws, size_t ws_size,
                              hipStream_t stream) {
    (void)in_sizes; (void)n_in; (void)out_size; (void)ws_size;
    const float* Y     = (const float*)d_in[0];   // [B,S,H]
    const float* maskY = (const float*)d_in[1];   // [B,S]
    const float* W     = (const float*)d_in[2];   // [H,H]
    const float* bv    = (const float*)d_in[3];   // [H]
    const float* wv    = (const float*)d_in[4];   // [H]
    float* out = (float*)d_out;                   // [B,H]

    char* ws = (char*)d_ws;
    unsigned short* Wt = (unsigned short*)ws;               // 512 KB
    float* scores = (float*)(ws + 524288);                  // 256 KB
    float* part   = (float*)(ws + 786432);                  // 1 MB

    convW<<<(H_ * H_) / 256, 256, 0, stream>>>(W, Wt);
    scores_kernel<<<(B_ * S_ / 16) / 2, 64, 0, stream>>>(Y, maskY, Wt, bv, wv, scores);
    softmax_kernel<<<B_, 256, 0, stream>>>(scores);
    wsum_partial<<<B_ * 16, 256, 0, stream>>>(Y, scores, part);
    wsum_reduce<<<(B_ * H_) / 256, 256, 0, stream>>>(part, out);
}
